// LinearTransformer_8117488189603
// MI455X (gfx1250) — matrix operand order repaired and checked by reading, not compiled, not run on hardware
//
#include <hip/hip_runtime.h>
#include <hip/hip_bf16.h>

// ---------------- problem constants (match reference) ----------------
#define HIDN    512
#define NLAYERS 4
#define NHEAD   4
#define DHEAD   128
#define FFND    2048
#define BBATCH  32
#define TSEQ    60
#define MROWS   (BBATCH * TSEQ)   // 1920, multiple of 16

typedef __attribute__((ext_vector_type(16))) __bf16 v16bf;
typedef __attribute__((ext_vector_type(8)))  float  v8f;

// ---------------- helpers ----------------
__device__ __forceinline__ unsigned short f2bf(float f) {
    unsigned int u = __float_as_uint(f);
    u += 0x7fffu + ((u >> 16) & 1u);   // round-to-nearest-even
    return (unsigned short)(u >> 16);
}

union V16U { uint4 u[2]; v16bf v; };

// A operand: 16x32 bf16 tile (M x K). ISA layout:
//  lanes 0-15 : row = lane,    v0..3 <- K=k0+0..7,  v4..7 <- K=k0+16..23
//  lanes 16-31: row = lane-16, v0..3 <- K=k0+8..15, v4..7 <- K=k0+24..31
__device__ __forceinline__ v16bf load_a(const unsigned short* A, int lda,
                                        int m0, int k0, int lane) {
    int r = lane & 15, hi = lane >> 4;
    const unsigned short* p = A + (size_t)(m0 + r) * lda + k0 + hi * 8;
    V16U u;
    u.u[0] = *reinterpret_cast<const uint4*>(p);
    u.u[1] = *reinterpret_cast<const uint4*>(p + 16);
    return u.v;
}

// B operand: 32x16 bf16 tile (K x N), weights stored transposed Wt[N][K]:
//  lanes 0-15 : col = lane,    v0..3 <- K=k0+0..7,  v4..7 <- K=k0+16..23
//  lanes 16-31: col = lane-16, v0..3 <- K=k0+8..15, v4..7 <- K=k0+24..31
__device__ __forceinline__ v16bf load_b(const unsigned short* Wt, int ldw,
                                        int n0, int k0, int lane) {
    int r = lane & 15, hi = lane >> 4;
    const unsigned short* p = Wt + (size_t)(n0 + r) * ldw + k0 + hi * 8;
    V16U u;
    u.u[0] = *reinterpret_cast<const uint4*>(p);
    u.u[1] = *reinterpret_cast<const uint4*>(p + 16);
    return u.v;
}

// ---------------- conversion kernels ----------------
__global__ void cvt_bf16(const float* __restrict__ in, unsigned short* __restrict__ out, int n) {
    int i = blockIdx.x * blockDim.x + threadIdx.x;
    if (i < n) out[i] = f2bf(in[i]);
}

// W is (K x N) row-major f32 -> Wt is (N x K) row-major bf16
__global__ void cvt_transpose_bf16(const float* __restrict__ W, unsigned short* __restrict__ Wt,
                                   int K, int N) {
    int i = blockIdx.x * blockDim.x + threadIdx.x;
    if (i >= K * N) return;
    int k = i / N, n = i - k * N;
    Wt[(size_t)n * K + k] = f2bf(W[i]);
}

// ---------------- WMMA GEMM ----------------
// C[M x N] = act(A_bf16[M x K] @ W[K x N] + bias[N]); W given transposed bf16 (N x K).
// Each wave computes a 16x64 strip: A operand loaded once per k-step, reused by
// 4 accumulators (4x WMMA density per A load; operand bytes/WMMA 64B -> 40B).
// ACT: 0 = none, 1 = elu(x)+1, 2 = exact gelu. OUTBF: write bf16 instead of f32.
template <int ACT, bool OUTBF>
__global__ __launch_bounds__(128) void wmma_gemm(const unsigned short* __restrict__ A,
                                                 const unsigned short* __restrict__ Wt,
                                                 const float* __restrict__ bias,
                                                 void* __restrict__ Cout,
                                                 int Mi, int N, int Kd) {
    int wave = threadIdx.x >> 5;
    int lane = threadIdx.x & 31;
    int strip = blockIdx.x * 4 + wave;  // one 16x64 strip per wave
    int nS = N >> 6;                    // strips per 16-row band (N % 64 == 0)
    int total = (Mi >> 4) * nS;
    if (strip >= total) return;
    int m0 = (strip / nS) << 4;
    int n0 = (strip % nS) << 6;

    v8f acc[4];
#pragma unroll
    for (int s = 0; s < 4; ++s) acc[s] = (v8f){0.f, 0.f, 0.f, 0.f, 0.f, 0.f, 0.f, 0.f};

#pragma unroll 2
    for (int k0 = 0; k0 < Kd; k0 += 32) {
        v16bf a = load_a(A, Kd, m0, k0, lane);
#pragma unroll
        for (int s = 0; s < 4; ++s) {
            v16bf b = load_b(Wt, Kd, n0 + s * 16, k0, lane);
            // (neg_a, A, neg_b, B, c_mod, C, reuse_a, reuse_b)
            acc[s] = __builtin_amdgcn_wmma_f32_16x16x32_bf16(false, a, false, b, (short)0,
                                                             acc[s], false, false);
        }
    }

    int r = lane & 15, hi = lane >> 4;
#pragma unroll
    for (int s = 0; s < 4; ++s) {
        int col = n0 + s * 16 + r;
        float bs = bias[col];
#pragma unroll
        for (int j = 0; j < 8; ++j) {
            int row = m0 + hi * 8 + j;  // C/D layout: VGPR j holds rows j and j+8
            float v = acc[s][j] + bs;
            if (ACT == 1) v = (v > 0.f) ? v + 1.f : __expf(v);           // elu(x)+1
            else if (ACT == 2) v = 0.5f * v * (1.f + erff(v * 0.70710678118654752f));
            if (OUTBF) ((unsigned short*)Cout)[(size_t)row * N + col] = f2bf(v);
            else       ((float*)Cout)[(size_t)row * N + col] = v;
        }
    }
}

// ---------------- causal linear-attention scan ----------------
// One workgroup per (batch, head). State S[d][m] (128x128 f32) lives in LDS.
// Thread m owns column m of S; per step: S[:,m] += k * v_m, acc = q . S[:,m].
__global__ __launch_bounds__(128) void attn_scan(const float* __restrict__ Q,
                                                 const float* __restrict__ K,
                                                 const float* __restrict__ V,
                                                 float* __restrict__ O) {
    __shared__ float S[DHEAD * DHEAD];     // 64 KB (WGP LDS is 320 KB)
    __shared__ float qs[DHEAD], ks[DHEAD];
    __shared__ float wred[4];
    int b = blockIdx.x >> 2, hh = blockIdx.x & 3;
    int m = threadIdx.x;                   // 0..127
    for (int i = m; i < DHEAD * DHEAD; i += DHEAD) S[i] = 0.f;
    float kc = 0.f;                        // cumulative K for own dim d = m
    __syncthreads();
    for (int t = 0; t < TSEQ; ++t) {
        size_t base = ((size_t)(b * TSEQ + t)) * HIDN + (size_t)hh * DHEAD;
        float q = Q[base + m], k = K[base + m], v = V[base + m];
        kc += k;
        qs[m] = q; ks[m] = k;
        float part = q * kc;               // Z partial (inclusive cumsum)
        for (int o = 16; o; o >>= 1) part += __shfl_down(part, o, 32);
        if ((m & 31) == 0) wred[m >> 5] = part;
        __syncthreads();
        float zinv = 1.f / (wred[0] + wred[1] + wred[2] + wred[3] + 1e-6f);
        float acc = 0.f;
#pragma unroll 4
        for (int d = 0; d < DHEAD; ++d) {  // conflict-free: thread m -> bank m
            float s = S[d * DHEAD + m] + ks[d] * v;
            S[d * DHEAD + m] = s;
            acc += qs[d] * s;
        }
        O[base + m] = acc * zinv;
        __syncthreads();
    }
}

// ---------------- residual + LayerNorm (in place on h) ----------------
__global__ __launch_bounds__(128) void ln_residual(float* __restrict__ h,
                                                   const float* __restrict__ dlt,
                                                   const float* __restrict__ g,
                                                   const float* __restrict__ be) {
    __shared__ float sm[4];
    int row = blockIdx.x, tid = threadIdx.x;
    float x[4];
    float s = 0.f;
#pragma unroll
    for (int j = 0; j < 4; ++j) {
        int c = tid + j * 128;
        x[j] = h[(size_t)row * HIDN + c] + dlt[(size_t)row * HIDN + c];
        s += x[j];
    }
    for (int o = 16; o; o >>= 1) s += __shfl_down(s, o, 32);
    if ((tid & 31) == 0) sm[tid >> 5] = s;
    __syncthreads();
    float mean = (sm[0] + sm[1] + sm[2] + sm[3]) * (1.f / HIDN);
    __syncthreads();
    float vv = 0.f;
#pragma unroll
    for (int j = 0; j < 4; ++j) { float t = x[j] - mean; vv += t * t; }
    for (int o = 16; o; o >>= 1) vv += __shfl_down(vv, o, 32);
    if ((tid & 31) == 0) sm[tid >> 5] = vv;
    __syncthreads();
    float rstd = rsqrtf((sm[0] + sm[1] + sm[2] + sm[3]) * (1.f / HIDN) + 1e-5f);
#pragma unroll
    for (int j = 0; j < 4; ++j) {
        int c = tid + j * 128;
        h[(size_t)row * HIDN + c] = (x[j] - mean) * rstd * g[c] + be[c];
    }
}

// ---------------- decode: z[b] = h[b, T-1, :] . dec_w + dec_b ----------------
__global__ __launch_bounds__(128) void decode_k(const float* __restrict__ h,
                                                const float* __restrict__ w,
                                                const float* __restrict__ b0,
                                                float* __restrict__ out) {
    __shared__ float sm[4];
    int b = blockIdx.x, tid = threadIdx.x;
    const float* row = h + ((size_t)b * TSEQ + (TSEQ - 1)) * HIDN;
    float s = 0.f;
    for (int i = tid; i < HIDN; i += 128) s += row[i] * w[i];
    for (int o = 16; o; o >>= 1) s += __shfl_down(s, o, 32);
    if ((tid & 31) == 0) sm[tid >> 5] = s;
    __syncthreads();
    if (tid == 0) out[b] = sm[0] + sm[1] + sm[2] + sm[3] + b0[0];
}

// ---------------- host orchestration ----------------
extern "C" void kernel_launch(void* const* d_in, const int* in_sizes, int n_in,
                              void* d_out, int out_size, void* d_ws, size_t ws_size,
                              hipStream_t stream) {
    (void)in_sizes; (void)n_in; (void)out_size; (void)ws_size;
    const float* x   = (const float*)d_in[0];
    // d_in[1]=p, d_in[2]=y : unused in forward pass
    const float* fcw = (const float*)d_in[3];
    const float* fcb = (const float*)d_in[4];
    const float* wq  = (const float*)d_in[5];
    const float* bq  = (const float*)d_in[6];
    const float* wk  = (const float*)d_in[7];
    const float* bk  = (const float*)d_in[8];
    const float* wv  = (const float*)d_in[9];
    const float* bv  = (const float*)d_in[10];
    const float* w1  = (const float*)d_in[11];
    const float* b1  = (const float*)d_in[12];
    const float* w2  = (const float*)d_in[13];
    const float* b2  = (const float*)d_in[14];
    const float* g1  = (const float*)d_in[15];
    const float* be1 = (const float*)d_in[16];
    const float* g2  = (const float*)d_in[17];
    const float* be2 = (const float*)d_in[18];
    const float* dw  = (const float*)d_in[19];
    const float* db  = (const float*)d_in[20];
    float* out = (float*)d_out;

    // workspace layout (~58 MB total)
    char* ws = (char*)d_ws;
    size_t off = 0;
    auto take = [&](size_t bytes) -> char* {
        char* p = ws + off;
        off = (off + bytes + 255) & ~(size_t)255;
        return p;
    };
    const size_t nw  = (size_t)HIDN * HIDN;    // 512*512
    const size_t n1  = (size_t)HIDN * FFND;    // 512*2048
    const size_t na  = (size_t)MROWS * HIDN;   // 1920*512
    unsigned short* wfc_t = (unsigned short*)take(nw * 2);
    unsigned short* wq_t  = (unsigned short*)take(NLAYERS * nw * 2);
    unsigned short* wk_t  = (unsigned short*)take(NLAYERS * nw * 2);
    unsigned short* wv_t  = (unsigned short*)take(NLAYERS * nw * 2);
    unsigned short* w1_t  = (unsigned short*)take(NLAYERS * n1 * 2);
    unsigned short* w2_t  = (unsigned short*)take(NLAYERS * n1 * 2);
    unsigned short* ab    = (unsigned short*)take(na * 2);                   // bf16 activations
    unsigned short* fb    = (unsigned short*)take((size_t)MROWS * FFND * 2); // bf16 FFN mid
    float* h    = (float*)take(na * 4);
    float* Qf   = (float*)take(na * 4);
    float* Kf   = (float*)take(na * 4);
    float* Vf   = (float*)take(na * 4);
    float* attn = (float*)take(na * 4);
    float* f2   = (float*)take(na * 4);

    dim3 cb(256), gb(128);
    auto gemm_grid = [](int M_, int N_) { return dim3(((M_ / 16) * (N_ / 64) + 3) / 4); };

    // --- one-time (per launch) weight conversion: f32 -> transposed bf16 ---
    cvt_transpose_bf16<<<(nw + 255) / 256, cb, 0, stream>>>(fcw, wfc_t, HIDN, HIDN);
    for (int i = 0; i < NLAYERS; ++i) {
        cvt_transpose_bf16<<<(nw + 255) / 256, cb, 0, stream>>>(wq + i * nw, wq_t + i * nw, HIDN, HIDN);
        cvt_transpose_bf16<<<(nw + 255) / 256, cb, 0, stream>>>(wk + i * nw, wk_t + i * nw, HIDN, HIDN);
        cvt_transpose_bf16<<<(nw + 255) / 256, cb, 0, stream>>>(wv + i * nw, wv_t + i * nw, HIDN, HIDN);
        cvt_transpose_bf16<<<(n1 + 255) / 256, cb, 0, stream>>>(w1 + i * n1, w1_t + i * n1, HIDN, FFND);
        cvt_transpose_bf16<<<(n1 + 255) / 256, cb, 0, stream>>>(w2 + i * n1, w2_t + i * n1, FFND, HIDN);
    }

    // --- input projection: h = x @ fc_w + fc_b ---
    cvt_bf16<<<((int)na + 255) / 256, cb, 0, stream>>>(x, ab, (int)na);
    wmma_gemm<0, false><<<gemm_grid(MROWS, HIDN), gb, 0, stream>>>(ab, wfc_t, fcb, h,
                                                                   MROWS, HIDN, HIDN);

    // --- transformer layers ---
    for (int i = 0; i < NLAYERS; ++i) {
        cvt_bf16<<<((int)na + 255) / 256, cb, 0, stream>>>(h, ab, (int)na);
        wmma_gemm<1, false><<<gemm_grid(MROWS, HIDN), gb, 0, stream>>>(
            ab, wq_t + i * nw, bq + (size_t)i * HIDN, Qf, MROWS, HIDN, HIDN);
        wmma_gemm<1, false><<<gemm_grid(MROWS, HIDN), gb, 0, stream>>>(
            ab, wk_t + i * nw, bk + (size_t)i * HIDN, Kf, MROWS, HIDN, HIDN);
        wmma_gemm<0, false><<<gemm_grid(MROWS, HIDN), gb, 0, stream>>>(
            ab, wv_t + i * nw, bv + (size_t)i * HIDN, Vf, MROWS, HIDN, HIDN);

        attn_scan<<<dim3(BBATCH * NHEAD), gb, 0, stream>>>(Qf, Kf, Vf, attn);
        ln_residual<<<dim3(MROWS), gb, 0, stream>>>(h, attn, g1 + (size_t)i * HIDN,
                                                    be1 + (size_t)i * HIDN);

        cvt_bf16<<<((int)na + 255) / 256, cb, 0, stream>>>(h, ab, (int)na);
        wmma_gemm<2, true><<<gemm_grid(MROWS, FFND), gb, 0, stream>>>(
            ab, w1_t + i * n1, b1 + (size_t)i * FFND, fb, MROWS, FFND, HIDN);
        wmma_gemm<0, false><<<gemm_grid(MROWS, HIDN), gb, 0, stream>>>(
            fb, w2_t + i * n1, b2 + (size_t)i * HIDN, f2, MROWS, HIDN, FFND);
        ln_residual<<<dim3(MROWS), gb, 0, stream>>>(h, f2, g2 + (size_t)i * HIDN,
                                                    be2 + (size_t)i * HIDN);
    }

    // --- decode head ---
    decode_k<<<dim3(BBATCH), gb, 0, stream>>>(h, dw, db, out);
}
